// FNO_lstm_38671885533647
// MI455X (gfx1250) — matrix operand order repaired and checked by reading, not compiled, not run on hardware
//
#include <hip/hip_runtime.h>
#include <math.h>

typedef __attribute__((ext_vector_type(16))) _Float16 v16h;
typedef __attribute__((ext_vector_type(8)))  float    v8f;

#define NN    16384      // B*T samples
#define S36   36
#define C66   66
#define FDIM  2376       // 66*36
#define KPF   2400       // FDIM padded to mult of 32
#define KPC   96         // 66  padded to mult of 32
#define NPC   80         // 66  padded to mult of 16 (5 col tiles, exact for NTW=5)
#define NPF2  2432       // 2376 padded to mult of 64 (152 col tiles, exact for NTW=4)
#define SPEC_SCALE 256.0f
#define EPS_IN 1e-5f

#define EP_BIAS 1
#define EP_RES  2
#define EP_GELU 4

__device__ __forceinline__ float gelu_f(float x) {
  return 0.5f * x * (1.0f + erff(x * 0.70710678118654752440f));
}
__device__ __forceinline__ float sigm_f(float x) {
  return 1.0f / (1.0f + expf(-x));
}

// cos/sin of 2*pi*d/6
__device__ const float C6tab[6] = {1.0f, 0.5f, -0.5f, -1.0f, -0.5f, 0.5f};
__device__ const float S6tab[6] = {0.0f, 0.86602540378443864676f, 0.86602540378443864676f,
                                   0.0f, -0.86602540378443864676f, -0.86602540378443864676f};

union HV { v16h v; uint4 u[2]; };

// ---------------------------------------------------------------------------
// Branchless WMMA GEMM: C[M x ncols](f32) = A[M x lda](f16 row-major, K-pad32)
//                                         x Bt[ncolsPad x ldb](f16 col-major)
// Each wave computes a (MTW*16) x (NTW*16) strip with NO guards in the k-loop
// (grids sized exactly, B padded to NTW*16 cols) so EXEC stays all-ones for
// every v_wmma. Epilogue: bias / residual-add / exact-GELU.
// ---------------------------------------------------------------------------
template <int MTW, int NTW>
__global__ void __launch_bounds__(128)
wmma_gemm(const _Float16* __restrict__ A, int lda,
          const _Float16* __restrict__ Bt, int ldb,
          float* __restrict__ C, int ldc,
          const float* __restrict__ bias,
          const float* __restrict__ resid,
          int ngroups, int ktiles, int ncols, int ep)
{
  const int wave = blockIdx.x * (blockDim.x >> 5) + (threadIdx.x >> 5);
  const int mt = wave / ngroups;          // row-strip index
  const int ng = wave % ngroups;          // col-strip index
  const int lane = threadIdx.x & 31;
  const int r  = lane & 15;
  const int hi = lane >> 4;

  v8f acc[MTW][NTW] = {};
  const _Float16* aRow[MTW];
  #pragma unroll
  for (int m = 0; m < MTW; ++m)
    aRow[m] = A + (size_t)((mt * MTW + m) * 16 + r) * lda + hi * 8;
  const _Float16* bCol0 = Bt + (size_t)(ng * (NTW * 16) + r) * ldb + hi * 8;

  for (int kt = 0; kt < ktiles; ++kt) {
    const int k0 = kt * 32;
    HV a[MTW];
    #pragma unroll
    for (int m = 0; m < MTW; ++m) {
      a[m].u[0] = *(const uint4*)(aRow[m] + k0);       // K k0+hi*8    ..+7
      a[m].u[1] = *(const uint4*)(aRow[m] + k0 + 16);  // K k0+16+hi*8 ..+7
    }
    __builtin_prefetch(aRow[0] + k0 + 32, 0, 1);       // speculative, safe
    #pragma unroll
    for (int j = 0; j < NTW; ++j) {
      HV b;
      const _Float16* bp = bCol0 + (size_t)j * 16 * ldb + k0;
      b.u[0] = *(const uint4*)(bp);                    // K k0+hi*8    ..+7
      b.u[1] = *(const uint4*)(bp + 16);               // K k0+16+hi*8 ..+7
      #pragma unroll
      for (int m = 0; m < MTW; ++m)
        acc[m][j] = __builtin_amdgcn_wmma_f32_16x16x32_f16(
            false, a[m].v, false, b.v, (short)0, acc[m][j], false, false);
    }
  }

  #pragma unroll
  for (int j = 0; j < NTW; ++j) {
    const int col = (ng * NTW + j) * 16 + r;
    if (col < ncols) {                                 // per-lane, after all wmma
      const float badd = (ep & EP_BIAS) ? bias[col] : 0.0f;
      #pragma unroll
      for (int m = 0; m < MTW; ++m) {
        #pragma unroll
        for (int v = 0; v < 8; ++v) {
          const int row = (mt * MTW + m) * 16 + hi * 8 + v;
          float x = acc[m][j][v] + badd;
          if (ep & EP_RES)  x += resid[(size_t)row * ldc + col];
          if (ep & EP_GELU) x = gelu_f(x);
          C[(size_t)row * ldc + col] = x;
        }
      }
    }
  }
}

// ---------------------------------------------------------------------------
// Lift: h[n,s,c] = (X[n,s]+0.01)*pw[0,c] + gx*pw[1,c] + gy*pw[2,c] + pb[c]
// ---------------------------------------------------------------------------
__global__ void lift_kernel(const float* __restrict__ X, const float* __restrict__ pw,
                            const float* __restrict__ pb, float* __restrict__ h, int total)
{
  int i = blockIdx.x * blockDim.x + threadIdx.x;
  if (i >= total) return;
  int c  = i % C66;
  int ns = i / C66;
  int s  = ns % S36;
  float xv = X[ns] + 0.01f;
  float gx = (float)(s / 6) * 0.2f;   // linspace(0,1,6)
  float gy = (float)(s % 6) * 0.2f;
  h[i] = xv * pw[c] + gx * pw[C66 + c] + gy * pw[2 * C66 + c] + pb[c];
}

// ---------------------------------------------------------------------------
// InstanceNorm (optional) + optional GELU + pack f32 [n][s][c] -> f16 A buffer
// mode 0: A[n][KPF] with k = s*66+c (tail zero padded)
// mode 1: A[(n*36+s)][KPC] (cols 66..95 zero padded)
// ---------------------------------------------------------------------------
__global__ void __launch_bounds__(128)
norm_pack(const float* __restrict__ in, _Float16* __restrict__ out,
          int mode, int do_norm, int do_gelu, float inv_scale)
{
  const int n = blockIdx.x;
  const int t = threadIdx.x;
  const float* row = in + (size_t)n * FDIM;
  if (t < C66) {
    float vals[S36];
    float s = 0.f, s2 = 0.f;
    #pragma unroll
    for (int i = 0; i < S36; ++i) {
      float x = row[i * C66 + t] * inv_scale;
      vals[i] = x; s += x; s2 += x * x;
    }
    float mean = 0.f, rstd = 1.f;
    if (do_norm) {
      mean = s * (1.0f / 36.0f);
      float var = s2 * (1.0f / 36.0f) - mean * mean;
      rstd = rsqrtf(var + EPS_IN);
    }
    #pragma unroll
    for (int i = 0; i < S36; ++i) {
      float x = (vals[i] - mean) * rstd;
      if (do_gelu) x = gelu_f(x);
      size_t o = (mode == 0) ? ((size_t)n * KPF + i * C66 + t)
                             : (((size_t)n * S36 + i) * KPC + t);
      out[o] = (_Float16)x;
    }
  } else if (mode == 0) {
    if (t < C66 + (KPF - FDIM))
      out[(size_t)n * KPF + FDIM + (t - C66)] = (_Float16)0.f;
  } else {
    if (t < KPC)
      for (int i = 0; i < S36; ++i)
        out[((size_t)n * S36 + i) * KPC + t] = (_Float16)0.f;
  }
}

// ---------------------------------------------------------------------------
// Build fused spectral operator Bt[j=(s',co)][k=(s,ci)] (col-major f16):
//   Bt = sum_m (DFT_r W_r - DFT_i W_i) C_r + (DFT_r W_i + DFT_i W_r) C_i
// Forward DFT (rfft2, 18 modes kx=0..5, ky=0..2), inverse per irfft2 semantics
// (ky=0 imag dropped, ky=1,2 Hermitian-doubled). Scaled by SPEC_SCALE for f16.
// ---------------------------------------------------------------------------
__global__ void __launch_bounds__(128)
build_spec(const float* __restrict__ w1r, const float* __restrict__ w1i,
           const float* __restrict__ w2r, const float* __restrict__ w2i,
           _Float16* __restrict__ Bt)
{
  const int j = blockIdx.x;                 // 0..NPF2-1
  _Float16* out = Bt + (size_t)j * KPF;
  if (j >= FDIM) {                          // zero padded columns
    for (int k = threadIdx.x; k < KPF; k += blockDim.x) out[k] = (_Float16)0.f;
    return;
  }
  const int sp = j / C66, co = j % C66;
  const int spx = sp / 6, spy = sp % 6;
  __shared__ float Cr[18], Ci[18];
  if (threadIdx.x < 18) {
    int kx = threadIdx.x / 3, ky = threadIdx.x % 3;
    int d = (kx * spx + ky * spy) % 6;
    float w = (ky == 0) ? (1.0f / 36.0f) : (2.0f / 36.0f);
    Cr[threadIdx.x] =  w * C6tab[d];
    Ci[threadIdx.x] = -w * S6tab[d];
  }
  __syncthreads();
  for (int k = threadIdx.x; k < KPF; k += blockDim.x) {
    float acc = 0.f;
    if (k < FDIM) {
      int s = k / C66, ci = k % C66;
      int sx = s / 6, sy = s % 6;
      const int base = (ci * C66 + co) * 9;
      #pragma unroll
      for (int m = 0; m < 18; ++m) {
        int kx = m / 3, ky = m % 3;
        int d = (kx * sx + ky * sy) % 6;
        float dr = C6tab[d], di = -S6tab[d];          // e^{-i theta}
        float wr, wi;
        if (kx < 3) { wr = w1r[base + kx * 3 + ky];       wi = w1i[base + kx * 3 + ky]; }
        else        { wr = w2r[base + (kx - 3) * 3 + ky]; wi = w2i[base + (kx - 3) * 3 + ky]; }
        acc += (dr * wr - di * wi) * Cr[m] + (dr * wi + di * wr) * Ci[m];
      }
      acc *= SPEC_SCALE;
    }
    out[k] = (_Float16)acc;
  }
}

// 66x66 conv weight [co][ci] -> f16 Bt [NPC][KPC] (already col-major for GEMM)
__global__ void pack_w66(const float* __restrict__ w, _Float16* __restrict__ Bt)
{
  int j = blockIdx.x;   // 0..79
  int k = threadIdx.x;  // 0..95
  Bt[j * KPC + k] = (j < C66 && k < C66) ? (_Float16)w[j * C66 + k] : (_Float16)0.f;
}

// gru0_wi [48][2376 (c-major: c*36+s)] -> Bt [48][KPF] with k = s*66+c
__global__ void pack_gru_wi(const float* __restrict__ wi, _Float16* __restrict__ Bt)
{
  int g = blockIdx.x;   // 0..47
  for (int k = threadIdx.x; k < KPF; k += blockDim.x) {
    float v = 0.f;
    if (k < FDIM) {
      int s = k / C66, c = k % C66;
      v = wi[g * FDIM + c * S36 + s];
    }
    Bt[(size_t)g * KPF + k] = (_Float16)v;
  }
}

// ---------------------------------------------------------------------------
// Fused 2-layer GRU recurrence + output head. One block per batch element.
// gi0 = precomputed x @ wi0.T + bi0 (from WMMA GEMM).
// ---------------------------------------------------------------------------
__global__ void __launch_bounds__(64)
gru_kernel(const float* __restrict__ gi0,
           const float* __restrict__ wh0, const float* __restrict__ bh0,
           const float* __restrict__ wi1, const float* __restrict__ bi1,
           const float* __restrict__ wh1, const float* __restrict__ bh1,
           const float* __restrict__ outw, const float* __restrict__ outb,
           float* __restrict__ y)
{
  __shared__ float swh0[768], swi1[768], swh1[768];
  __shared__ float sbh0[48], sbi1[48], sbh1[48], sow[16];
  __shared__ float h0s[16], h1s[16], h0n[16];
  __shared__ float g0[48], g1i[48], g1h[48];
  const int b = blockIdx.x;
  const int t = threadIdx.x;
  for (int i = t; i < 768; i += 64) { swh0[i] = wh0[i]; swi1[i] = wi1[i]; swh1[i] = wh1[i]; }
  if (t < 48) { sbh0[t] = bh0[t]; sbi1[t] = bi1[t]; sbh1[t] = bh1[t]; }
  if (t < 16) { sow[t] = outw[t]; h0s[t] = 0.f; h1s[t] = 0.f; }
  __syncthreads();
  for (int step = 0; step < 256; ++step) {
    const int n = b * 256 + step;
    const float* gi = gi0 + (size_t)n * 48;
    if (t < 48) {
      float a = sbh0[t];
      #pragma unroll
      for (int j = 0; j < 16; ++j) a += h0s[j] * swh0[t * 16 + j];
      g0[t] = a;
    }
    __syncthreads();
    if (t < 16) {
      float rr = sigm_f(gi[t]      + g0[t]);
      float zz = sigm_f(gi[16 + t] + g0[16 + t]);
      float nn = tanhf (gi[32 + t] + rr * g0[32 + t]);
      h0n[t] = (1.f - zz) * nn + zz * h0s[t];
    }
    __syncthreads();
    if (t < 48) {
      float a = sbi1[t], c = sbh1[t];
      #pragma unroll
      for (int j = 0; j < 16; ++j) { a += h0n[j] * swi1[t * 16 + j]; c += h1s[j] * swh1[t * 16 + j]; }
      g1i[t] = a; g1h[t] = c;
    }
    __syncthreads();
    if (t < 16) {
      float rr = sigm_f(g1i[t] + g1h[t]);
      float zz = sigm_f(g1i[16 + t] + g1h[16 + t]);
      float nn = tanhf (g1i[32 + t] + rr * g1h[32 + t]);
      h1s[t] = (1.f - zz) * nn + zz * h1s[t];
      h0s[t] = h0n[t];
    }
    __syncthreads();
    if (t == 0) {
      float acc = outb[0];
      for (int j = 0; j < 16; ++j) acc += h1s[j] * sow[j];
      y[n] = acc;
    }
    __syncthreads();
  }
}

extern "C" void kernel_launch(void* const* d_in, const int* in_sizes, int n_in,
                              void* d_out, int out_size, void* d_ws, size_t ws_size,
                              hipStream_t stream)
{
  const float* X   = (const float*)d_in[0];
  const float* p_w = (const float*)d_in[1];
  const float* p_b = (const float*)d_in[2];
  const float* c_w[2][4] = {
      {(const float*)d_in[3], (const float*)d_in[4], (const float*)d_in[5], (const float*)d_in[6]},
      {(const float*)d_in[7], (const float*)d_in[8], (const float*)d_in[9], (const float*)d_in[10]}};
  const float* m_w[2][2] = {{(const float*)d_in[11], (const float*)d_in[13]},
                            {(const float*)d_in[15], (const float*)d_in[17]}};
  const float* m_b[2][2] = {{(const float*)d_in[12], (const float*)d_in[14]},
                            {(const float*)d_in[16], (const float*)d_in[18]}};
  const float* w_w[2] = {(const float*)d_in[19], (const float*)d_in[21]};
  const float* w_b[2] = {(const float*)d_in[20], (const float*)d_in[22]};
  const float* g0wi = (const float*)d_in[23];
  const float* g0wh = (const float*)d_in[24];
  const float* g0bi = (const float*)d_in[25];
  const float* g0bh = (const float*)d_in[26];
  const float* g1wi = (const float*)d_in[27];
  const float* g1wh = (const float*)d_in[28];
  const float* g1bi = (const float*)d_in[29];
  const float* g1bh = (const float*)d_in[30];
  const float* outw = (const float*)d_in[31];
  const float* outb = (const float*)d_in[32];
  float* y = (float*)d_out;

  size_t off = 0;
  auto take = [&](size_t bytes) -> void* {
    void* p = (char*)d_ws + off;
    off += (bytes + 255) & ~(size_t)255;
    return p;
  };
  float*    h    = (float*)take((size_t)NN * FDIM * 4);
  float*    tmp  = (float*)take((size_t)NN * FDIM * 4);
  _Float16* Abuf = (_Float16*)take((size_t)NN * S36 * KPC * 2); // covers both layouts
  _Float16* Bs[2];
  Bs[0] = (_Float16*)take((size_t)NPF2 * KPF * 2);
  Bs[1] = (_Float16*)take((size_t)NPF2 * KPF * 2);
  _Float16* Bw[2][3];
  for (int bl = 0; bl < 2; ++bl)
    for (int q = 0; q < 3; ++q)
      Bw[bl][q] = (_Float16*)take((size_t)NPC * KPC * 2);
  _Float16* BtG = (_Float16*)take((size_t)48 * KPF * 2);
  float*    gi0 = (float*)take((size_t)NN * 48 * 4);
  (void)ws_size; (void)n_in; (void)in_sizes; (void)out_size;

  // ---- prep: pack all weights into WMMA layouts (deterministic per launch)
  for (int bl = 0; bl < 2; ++bl) {
    build_spec<<<NPF2, 128, 0, stream>>>(c_w[bl][0], c_w[bl][1], c_w[bl][2], c_w[bl][3], Bs[bl]);
    pack_w66<<<NPC, KPC, 0, stream>>>(m_w[bl][0], Bw[bl][0]);
    pack_w66<<<NPC, KPC, 0, stream>>>(m_w[bl][1], Bw[bl][1]);
    pack_w66<<<NPC, KPC, 0, stream>>>(w_w[bl],    Bw[bl][2]);
  }
  pack_gru_wi<<<48, 128, 0, stream>>>(g0wi, BtG);

  // ---- lift
  {
    int total = NN * FDIM;
    lift_kernel<<<(total + 255) / 256, 256, 0, stream>>>(X, p_w, p_b, h, total);
  }

  // Exact grids: waves = (mtiles/MTW) * ngroups, 4 waves/block, no remainder.
  // spec: (1024/2)*38 = 19456 waves -> 4864 blocks   <MTW=2, NTW=4>
  // mlp : (36864/2)*1 = 18432 waves -> 4608 blocks   <MTW=2, NTW=5>
  // gru : (1024/2)*1  =   512 waves ->  128 blocks   <MTW=2, NTW=3>
  const int NG_SPEC = (NPF2 / 16) / 4;  // 38

  // ---- FNO blocks
  for (int bl = 0; bl < 2; ++bl) {
    // inorm1 -> A (mode 0: [n][2400])
    norm_pack<<<NN, 128, 0, stream>>>(h, Abuf, 0, 1, 0, 1.0f);
    // spectral: tmp = A x Bspec (scaled by SPEC_SCALE)
    wmma_gemm<2, 4><<<(NN / 32) * NG_SPEC / 4, 128, 0, stream>>>(
        Abuf, KPF, Bs[bl], KPF, tmp, FDIM, nullptr, nullptr,
        NG_SPEC, KPF / 32, FDIM, 0);
    // inorm2 (undo scale) -> A (mode 1: [(n,s)][96])
    norm_pack<<<NN, 128, 0, stream>>>(tmp, Abuf, 1, 1, 0, 1.0f / SPEC_SCALE);
    // mlp a: tmp = A x Wa + ba
    wmma_gemm<2, 5><<<(NN * S36 / 32) / 4, 128, 0, stream>>>(
        Abuf, KPC, Bw[bl][0], KPC, tmp, C66, m_b[bl][0], nullptr,
        1, KPC / 32, C66, EP_BIAS);
    // gelu + pack -> A (mode 1)
    norm_pack<<<NN, 128, 0, stream>>>(tmp, Abuf, 1, 0, 1, 1.0f);
    // mlp b: tmp = A x Wb + bb
    wmma_gemm<2, 5><<<(NN * S36 / 32) / 4, 128, 0, stream>>>(
        Abuf, KPC, Bw[bl][1], KPC, tmp, C66, m_b[bl][1], nullptr,
        1, KPC / 32, C66, EP_BIAS);
    // pack residual input h -> A (mode 1, plain)
    norm_pack<<<NN, 128, 0, stream>>>(h, Abuf, 1, 0, 0, 1.0f);
    // skip: h = gelu(tmp + A x Wskip + bskip)
    wmma_gemm<2, 5><<<(NN * S36 / 32) / 4, 128, 0, stream>>>(
        Abuf, KPC, Bw[bl][2], KPC, h, C66, w_b[bl], tmp,
        1, KPC / 32, C66, EP_BIAS | EP_RES | EP_GELU);
  }

  // ---- GRU input projection: gi0 = feats x wi0.T + bi0
  norm_pack<<<NN, 128, 0, stream>>>(h, Abuf, 0, 0, 0, 1.0f);
  wmma_gemm<2, 3><<<(NN / 32) / 4, 128, 0, stream>>>(
      Abuf, KPF, BtG, KPF, gi0, 48, g0bi, nullptr,
      1, KPF / 32, 48, EP_BIAS);

  // ---- fused GRU recurrence (2 layers) + head
  gru_kernel<<<64, 64, 0, stream>>>(gi0, g0wh, g0bh, g1wi, g1bi, g1wh, g1bh,
                                    outw, outb, y);
}
